// additive_attention_20916490731987
// MI455X (gfx1250) — matrix operand order repaired and checked by reading, not compiled, not run on hardware
//
#include <hip/hip_runtime.h>
#include <hip/hip_bf16.h>

// Problem constants (from reference)
#define B_   16
#define S_   2048
#define H_   1024
#define A_   1024
#define P_   64
#define L_   51
#define G_   1024
#define KX   1216   // H + P + P + 64 (L padded 51->64)

typedef __attribute__((ext_vector_type(16))) __bf16 v16bf;
typedef __attribute__((ext_vector_type(8)))  float  v8f;

union BF16x16 { uint4 u[2]; v16bf v; };

// ---------------------------------------------------------------------------
// Kernel 1: pack [W_h | W_s | W_o | W_l(padded)] into bf16 Wcat[A][KX]
// ---------------------------------------------------------------------------
__global__ void aa_pack_wcat(const float* __restrict__ Wh, const float* __restrict__ Ws,
                             const float* __restrict__ Wo, const float* __restrict__ Wl,
                             __bf16* __restrict__ Wcat) {
  int idx = blockIdx.x * 256 + threadIdx.x;            // < A_*KX
  if (idx >= A_ * KX) return;
  int a = idx / KX, k = idx % KX;
  float val;
  if (k < H_)              val = Wh[a * H_ + k];
  else if (k < H_ + P_)    val = Ws[a * P_ + (k - H_)];
  else if (k < H_ + 2*P_)  val = Wo[a * P_ + (k - H_ - P_)];
  else { int kl = k - H_ - 2*P_; val = (kl < L_) ? Wl[a * L_ + kl] : 0.0f; }
  Wcat[idx] = (__bf16)val;
}

// ---------------------------------------------------------------------------
// Kernel 2: bias[b][a] = q[b]*W_q[a] + g[b]*W_g[a]
// ---------------------------------------------------------------------------
__global__ void aa_bias(const float* __restrict__ q, const float* __restrict__ g,
                        const float* __restrict__ Wq, const float* __restrict__ Wg,
                        float* __restrict__ bias) {
  int idx = blockIdx.x * 256 + threadIdx.x;            // < B_*A_
  if (idx >= B_ * A_) return;
  int b = idx >> 10, a = idx & 1023;
  float s = 0.0f;
  for (int k = 0; k < H_; ++k) s += q[b * H_ + k] * Wq[a * H_ + k];
  for (int k = 0; k < G_; ++k) s += g[b * G_ + k] * Wg[a * G_ + k];
  bias[idx] = s;
}

// ---------------------------------------------------------------------------
// Kernel 3: fused logits. One workgroup = 16 s-rows of one batch.
// X row = [h(1024) | Ps_tab[ps](64) | Po_tab[po](64) | l(51)+pad] staged in LDS
// as bf16. 8 waves x 8 N-tiles each; K-outer / tile-inner; per K-step all 16
// B-fragment global_load_b128 issue into DISTINCT registers before the 8
// v_wmma_f32_16x16x32_bf16 drain with staggered loadcnt waits. Epilogue fuses
// +bias, tanh, dot with v, cross-lane reduce, masking.
// ---------------------------------------------------------------------------
__global__ __launch_bounds__(256) void aa_logits(
    const float* __restrict__ h, const int* __restrict__ ps,
    const int* __restrict__ po, const float* __restrict__ l,
    const unsigned char* __restrict__ mask,
    const float* __restrict__ Pstab, const float* __restrict__ Potab,
    const __bf16* __restrict__ Wcat, const float* __restrict__ bias,
    const float* __restrict__ vvec, float* __restrict__ logits) {
  __shared__ __align__(16) __bf16 Xs[16][KX + 8];   // +8 pad keeps 16B row align
  __shared__ float wsum[8][16];                     // per-wave row partials

  const int tid = threadIdx.x;
  const int b   = blockIdx.x >> 7;                  // / (S_/16 = 128)
  const int s0  = (blockIdx.x & 127) << 4;

  // --- stage activation tile (16 rows x KX) as bf16 ---
  {
    int r  = tid >> 4;                              // 0..15 row
    int c0 = tid & 15;
    int row = b * S_ + s0 + r;
    const float* hrow = h + (size_t)row * H_;
    const float* lrow = l + (size_t)row * L_;
    const float* pse  = Pstab + (size_t)ps[row] * P_;
    const float* poe  = Potab + (size_t)po[row] * P_;
    for (int k = c0; k < KX; k += 16) {
      float val;
      if (k < H_)              val = hrow[k];
      else if (k < H_ + P_)    val = pse[k - H_];
      else if (k < H_ + 2*P_)  val = poe[k - H_ - P_];
      else { int kl = k - H_ - 2*P_; val = (kl < L_) ? lrow[kl] : 0.0f; }
      Xs[r][k] = (__bf16)val;
    }
  }

  const int lane = tid & 31;
  const int wave = tid >> 5;
  const int m16  = lane & 15;                       // tile col / A-matrix row
  const int kh   = lane >> 4;                       // K half selector

  // hoist epilogue scalars (bias, v) while the LDS stage settles
  float bv[8], vv[8];
#pragma unroll
  for (int t = 0; t < 8; ++t) {
    const int ncol = wave * 128 + t * 16 + m16;
    bv[t] = bias[b * A_ + ncol];
    vv[t] = vvec[ncol];
  }

  __syncthreads();

  // 8 accumulator tiles per wave (N = wave*128 + t*16 .. +15)
  v8f acc[8];
#pragma unroll
  for (int t = 0; t < 8; ++t) acc[t] = (v8f){};

  const __bf16* xrow  = &Xs[m16][0];                            // A row in LDS
  const __bf16* wbase = Wcat + (size_t)(wave * 128 + m16) * KX; // B col base

  // K-outer, tile-inner: one LDS A-fragment -> 8 WMMAs
  for (int kk = 0; kk < KX; kk += 32) {
    BF16x16 af;
    // A 16x32 bf16 layout: lanes0-15 K kk..kk+15; lanes16-31 K kk+16..kk+31
    af.u[0] = *(const uint4*)(xrow + kk + 16 * kh);
    af.u[1] = *(const uint4*)(xrow + kk + 16 * kh + 8);

    // issue ALL B-fragment loads into distinct registers first
    BF16x16 bf[8];
#pragma unroll
    for (int t = 0; t < 8; ++t) {
      // B 32x16 bf16 layout: lanes0-15 K kk..kk+15; lanes16-31 K kk+16..kk+31
      const __bf16* wp = wbase + (size_t)t * 16 * KX + kk + 16 * kh;
      bf[t].u[0] = *(const uint4*)(wp);
      bf[t].u[1] = *(const uint4*)(wp + 8);
    }
    if (kk + 32 < KX) {
      __builtin_prefetch(wbase + kk + 32 + 16 * kh, 0, 1);
      __builtin_prefetch(wbase + 4 * 16 * KX + kk + 32 + 16 * kh, 0, 1);
    }

    // drain: 8 WMMAs against the shared A fragment
#pragma unroll
    for (int t = 0; t < 8; ++t)
      acc[t] = __builtin_amdgcn_wmma_f32_16x16x32_bf16(
          false, af.v, false, bf[t].v, (short)0, acc[t], false, false);
  }

  // epilogue: + bias, tanh, * v, accumulate per-row partial logit
  float rsum[8];
#pragma unroll
  for (int i = 0; i < 8; ++i) rsum[i] = 0.0f;
#pragma unroll
  for (int t = 0; t < 8; ++t) {
#pragma unroll
    for (int i = 0; i < 8; ++i)                     // row = i + 8*kh, col = m16
      rsum[i] += tanhf(acc[t][i] + bv[t]) * vv[t];
  }

  // reduce across the 16 columns held by each half-wave
#pragma unroll
  for (int i = 0; i < 8; ++i) {
    float r = rsum[i];
    r += __shfl_xor(r, 1, 16);
    r += __shfl_xor(r, 2, 16);
    r += __shfl_xor(r, 4, 16);
    r += __shfl_xor(r, 8, 16);
    rsum[i] = r;
  }
  if (m16 == 0) {
    int rbase = kh * 8;
#pragma unroll
    for (int i = 0; i < 8; ++i) wsum[wave][rbase + i] = rsum[i];
  }
  __syncthreads();

  if (tid < 16) {                                   // deterministic 8-way sum
    float lg = 0.0f;
#pragma unroll
    for (int w = 0; w < 8; ++w) lg += wsum[w][tid];
    int s = s0 + tid;
    logits[b * S_ + s] = mask[b * S_ + s] ? -1e9f : lg;
  }
}

// ---------------------------------------------------------------------------
// Kernel 4: softmax over S per batch row
// ---------------------------------------------------------------------------
__global__ void aa_softmax(const float* __restrict__ logits, float* __restrict__ alpha) {
  __shared__ float red[256];
  const int b = blockIdx.x, tid = threadIdx.x;
  const float* row = logits + b * S_;

  float m = -3.4e38f;
  for (int s = tid; s < S_; s += 256) m = fmaxf(m, row[s]);
  red[tid] = m; __syncthreads();
  for (int off = 128; off > 0; off >>= 1) {
    if (tid < off) red[tid] = fmaxf(red[tid], red[tid + off]);
    __syncthreads();
  }
  const float mx = red[0];
  __syncthreads();

  float sum = 0.0f;
  for (int s = tid; s < S_; s += 256) {
    float e = __expf(row[s] - mx);
    alpha[b * S_ + s] = e;
    sum += e;
  }
  red[tid] = sum; __syncthreads();
  for (int off = 128; off > 0; off >>= 1) {
    if (tid < off) red[tid] += red[tid + off];
    __syncthreads();
  }
  const float inv = 1.0f / red[0];
  for (int s = tid; s < S_; s += 256) alpha[b * S_ + s] *= inv;
}

// ---------------------------------------------------------------------------
// Kernel 5: partial o = alpha^T h over S-chunks of 512 (coalesced over H)
// grid = B * 4(hc) * 4(sc)
// ---------------------------------------------------------------------------
__global__ void aa_wsum(const float* __restrict__ h, const float* __restrict__ alpha,
                        float* __restrict__ partial) {
  __shared__ float as[512];
  const int tid = threadIdx.x;
  const int b  = blockIdx.x >> 4;
  const int hc = (blockIdx.x >> 2) & 3;
  const int sc = blockIdx.x & 3;
  const int sbase = sc * 512;
  as[tid]       = alpha[b * S_ + sbase + tid];
  as[tid + 256] = alpha[b * S_ + sbase + tid + 256];
  __syncthreads();
  const int col = hc * 256 + tid;
  const float* hp = h + (size_t)(b * S_ + sbase) * H_ + col;
  float acc = 0.0f;
  for (int s = 0; s < 512; ++s) acc += as[s] * hp[(size_t)s * H_];
  partial[(sc * B_ + b) * H_ + col] = acc;
}

// ---------------------------------------------------------------------------
// Kernel 6: reduce 4 S-chunk partials -> o[B][H]
// ---------------------------------------------------------------------------
__global__ void aa_reduce(const float* __restrict__ partial, float* __restrict__ out) {
  int idx = blockIdx.x * 256 + threadIdx.x;          // < B_*H_
  if (idx >= B_ * H_) return;
  float s = 0.0f;
#pragma unroll
  for (int c = 0; c < 4; ++c) s += partial[c * (B_ * H_) + idx];
  out[idx] = s;
}

// ---------------------------------------------------------------------------
extern "C" void kernel_launch(void* const* d_in, const int* in_sizes, int n_in,
                              void* d_out, int out_size, void* d_ws, size_t ws_size,
                              hipStream_t stream) {
  (void)in_sizes; (void)n_in; (void)out_size; (void)ws_size;
  const float* h   = (const float*)d_in[0];
  const float* q   = (const float*)d_in[1];
  const int*   ps  = (const int*)d_in[2];
  const int*   po  = (const int*)d_in[3];
  const float* l   = (const float*)d_in[4];
  const float* g   = (const float*)d_in[5];
  const unsigned char* mask = (const unsigned char*)d_in[6];
  const float* Wh  = (const float*)d_in[7];
  const float* Wq  = (const float*)d_in[8];
  const float* Ws  = (const float*)d_in[9];
  const float* Wo  = (const float*)d_in[10];
  const float* Wl  = (const float*)d_in[11];
  const float* Wg  = (const float*)d_in[12];
  const float* v   = (const float*)d_in[13];
  const float* Pstab = (const float*)d_in[14];
  const float* Potab = (const float*)d_in[15];
  float* out = (float*)d_out;

  // workspace layout (bytes, 16B-aligned partitions)
  char* ws = (char*)d_ws;
  __bf16* Wcat   = (__bf16*)(ws + 0);          // 1024*1216*2 = 2,490,368
  float* bias    = (float*)(ws + 2490368);     // 16*1024*4   =    65,536
  float* logits  = (float*)(ws + 2555904);     // 16*2048*4   =   131,072
  float* alpha   = (float*)(ws + 2686976);     // 16*2048*4   =   131,072
  float* partial = (float*)(ws + 2818048);     // 4*16*1024*4 =   262,144

  aa_pack_wcat<<<dim3((A_ * KX + 255) / 256), dim3(256), 0, stream>>>(Wh, Ws, Wo, Wl, Wcat);
  aa_bias<<<dim3((B_ * A_ + 255) / 256), dim3(256), 0, stream>>>(q, g, Wq, Wg, bias);
  aa_logits<<<dim3(B_ * (S_ / 16)), dim3(256), 0, stream>>>(
      h, ps, po, l, mask, Pstab, Potab, Wcat, bias, v, logits);
  aa_softmax<<<dim3(B_), dim3(256), 0, stream>>>(logits, alpha);
  aa_wsum<<<dim3(B_ * 4 * 4), dim3(256), 0, stream>>>(h, alpha, partial);
  aa_reduce<<<dim3((B_ * H_ + 255) / 256), dim3(256), 0, stream>>>(partial, out);
}
